// DetectNet_71262097375252
// MI455X (gfx1250) — matrix operand order repaired and checked by reading, not compiled, not run on hardware
//
#include <hip/hip_runtime.h>
#include <hip/hip_bf16.h>
#include <math.h>

typedef __attribute__((ext_vector_type(16))) _Float16 v16h;
typedef __attribute__((ext_vector_type(8)))  _Float16 v8h;
typedef __attribute__((ext_vector_type(8)))  float    v8f;

// ---------------- dims ----------------
#define IMH 640
#define IMW 960
#define C1H 106
#define C1W 160
#define P1H 53
#define P1W 80
#define C2H 51
#define C2W 78
#define P2H 25
#define P2W 39
#define C3H 23
#define C3W 37
#define FLATK 108928   // 23*37*128
#define HID  256
#define GW_ 60
#define GH_ 40
#define NOUT 38400     // 40*60*16
#define NBOX 4800
#define NWORD 150      // 4800/32
#define NPAD 8192

__device__ __forceinline__ float sigmoidf(float v) { return 1.0f / (1.0f + __expf(-v)); }

// ---------------- init ----------------
__global__ void init_h1(float* h1) {
    int t = threadIdx.x;
    if (t < HID) h1[t] = 0.0f;
}

// ---------------- conv1: (640,960,3) -> (106,160,32), 6x6 stride 6, relu --------
__global__ void conv1_kernel(const float* __restrict__ img, const float* __restrict__ w,
                             const float* __restrict__ b, float* __restrict__ out) {
    int o = blockIdx.x * 256 + threadIdx.x;
    const int total = C1H * C1W * 32;
    if (o >= total) return;
    int oc = o & 31;
    int sp = o >> 5;
    int ox = sp % C1W, oy = sp / C1W;
    float acc = b[oc];
    for (int ky = 0; ky < 6; ++ky)
        for (int kx = 0; kx < 6; ++kx) {
            const float* ip = img + ((oy * 6 + ky) * IMW + (ox * 6 + kx)) * 3;
            const float* wp = w + ((ky * 6 + kx) * 3) * 32 + oc;
            acc += ip[0] * wp[0] + ip[1] * wp[32] + ip[2] * wp[64];
        }
    out[o] = fmaxf(acc, 0.0f);
}

// ---------------- maxpool 2x2 generic ----------------
__global__ void maxpool_kernel(const float* __restrict__ in, float* __restrict__ out,
                               int oh, int ow, int iw, int ch, int chShift) {
    int o = blockIdx.x * 256 + threadIdx.x;
    int total = oh * ow * ch;
    if (o >= total) return;
    int oc = o & (ch - 1);
    int sp = o >> chShift;
    int ox = sp % ow, oy = sp / ow;
    const float* p0 = in + ((2 * oy) * iw + 2 * ox) * ch + oc;
    const float* p1 = in + ((2 * oy + 1) * iw + 2 * ox) * ch + oc;
    float m = fmaxf(fmaxf(p0[0], p0[ch]), fmaxf(p1[0], p1[ch]));
    out[o] = m;
}

// ---------------- conv2: (53,80,32) -> (51,78,64), 3x3, relu ----------------
__global__ void conv2_kernel(const float* __restrict__ in, const float* __restrict__ w,
                             const float* __restrict__ b, float* __restrict__ out) {
    int o = blockIdx.x * 256 + threadIdx.x;
    const int total = C2H * C2W * 64;
    if (o >= total) return;
    int oc = o & 63;
    int sp = o >> 6;
    int ox = sp % C2W, oy = sp / C2W;
    float acc = b[oc];
    for (int ky = 0; ky < 3; ++ky)
        for (int kx = 0; kx < 3; ++kx) {
            const float* ip = in + ((oy + ky) * P1W + (ox + kx)) * 32;
            const float* wp = w + ((ky * 3 + kx) * 32) * 64 + oc;
            #pragma unroll
            for (int ci = 0; ci < 32; ++ci) acc += ip[ci] * wp[ci * 64];
        }
    out[o] = fmaxf(acc, 0.0f);
}

// ---------------- conv3: (25,39,64) -> (23,37,128), 3x3, relu, f16 out --------
// Output feeds only dense1 (WMMA f16 A-operand), so emit _Float16 directly.
__global__ void conv3_kernel(const float* __restrict__ in, const float* __restrict__ w,
                             const float* __restrict__ b, _Float16* __restrict__ out) {
    int o = blockIdx.x * 256 + threadIdx.x;
    const int total = C3H * C3W * 128;
    if (o >= total) return;
    int oc = o & 127;
    int sp = o >> 7;
    int ox = sp % C3W, oy = sp / C3W;
    float acc = b[oc];
    for (int ky = 0; ky < 3; ++ky)
        for (int kx = 0; kx < 3; ++kx) {
            const float* ip = in + ((oy + ky) * P2W + (ox + kx)) * 64;
            const float* wp = w + ((ky * 3 + kx) * 64) * 128 + oc;
            #pragma unroll
            for (int ci = 0; ci < 64; ++ci) acc += ip[ci] * wp[ci * 128];
        }
    out[o] = (_Float16)fmaxf(acc, 0.0f);
}

// ---------------- WMMA A-operand: 16x32 f16 tile, row M=0 = x[k..k+31] -------
// A operand in the contiguous K order this code uses (B matches; wave32): lane<16 holds K={0..15}; lane>=16 holds
// K={16..31}. Two contiguous b128 loads per lane; every lane loads the
// row-0 data (rows 1..15 of C become duplicates and are discarded).
__device__ __forceinline__ v16h load_rowvec_a(const _Float16* __restrict__ xh,
                                              int k, int lane) {
    const v8h* pa = reinterpret_cast<const v8h*>(xh + k + ((lane >> 4) << 4));
    v8h lo = pa[0];   // K base+0..7
    v8h hi = pa[1];   // K base+8..15
    return __builtin_shufflevector(lo, hi, 0, 1, 2, 3, 4, 5, 6, 7,
                                   8, 9, 10, 11, 12, 13, 14, 15);
}

// ---------------- dense1: x(1x108928) @ wd1(108928x256), split-K WMMA ----------
// grid (16 ntiles, 16 ksplits) x 256 thr (8 waves): 128 K-slices of 27 steps.
__global__ void dense1_wmma(const _Float16* __restrict__ xh, const float* __restrict__ wd1,
                            float* __restrict__ h1) {
    const int Ksteps = FLATK / 32;         // 3404
    const int stepsPer = 27;               // ceil(3404/128)
    int lane = threadIdx.x & 31;
    int wave = threadIdx.x >> 5;
    int n0 = blockIdx.x * 16;
    int sliceId = blockIdx.y * 8 + wave;   // 0..127
    int ks0 = sliceId * stepsPer;
    int nlane = lane & 15;
    int kOff = (lane >> 4) << 4;           // 0 or 16 (B-operand K half)
    v8f c = {};
    for (int s = 0; s < stepsPer; ++s) {
        int ks = ks0 + s;
        if (ks >= Ksteps) break;           // wave-uniform
        int k = ks << 5;
        v16h a = load_rowvec_a(xh, k, lane);
        const float* wp = wd1 + (size_t)(k + kOff) * HID + n0 + nlane;
        __builtin_prefetch(wp + (size_t)32 * HID, 0, 1);
        v16h b;
        #pragma unroll
        for (int h = 0; h < 16; ++h) b[h] = (_Float16)wp[(size_t)h * HID];
        c = __builtin_amdgcn_wmma_f32_16x16x32_f16(false, a, false, b, (short)0, c,
                                                   false, false);
    }
    if (lane < 16) atomicAdd(&h1[n0 + lane], c[0]);
}

// bias + relu; also emit f16 copy for dense2's WMMA A-operand
__global__ void bias_relu_h1(float* h1, const float* __restrict__ bd1,
                             _Float16* __restrict__ h1h) {
    int t = threadIdx.x;
    if (t < HID) {
        float v = fmaxf(h1[t] + bd1[t], 0.0f);
        h1[t] = v;
        h1h[t] = (_Float16)v;
    }
}

// ---------------- dense2: h(256) @ wd2(256x38400) + bd2, WMMA ----------------
// grid 300 x 256 thr: wave -> one 16-col N tile, full K (8 steps of 32).
__global__ void dense2_wmma(const _Float16* __restrict__ h1h, const float* __restrict__ wd2,
                            const float* __restrict__ bd2, float* __restrict__ y) {
    int lane = threadIdx.x & 31;
    int wave = threadIdx.x >> 5;
    int ntile = blockIdx.x * 8 + wave;     // 0..2399
    int n0 = ntile * 16;
    int nlane = lane & 15;
    int kOff = (lane >> 4) << 4;
    v8f c = {};
    #pragma unroll
    for (int ks = 0; ks < 8; ++ks) {
        int k = ks << 5;
        v16h a = load_rowvec_a(h1h, k, lane);
        const float* wp = wd2 + (size_t)(k + kOff) * NOUT + n0 + nlane;
        v16h b;
        #pragma unroll
        for (int h = 0; h < 16; ++h) b[h] = (_Float16)wp[(size_t)h * NOUT];
        c = __builtin_amdgcn_wmma_f32_16x16x32_f16(false, a, false, b, (short)0, c,
                                                   false, false);
    }
    if (lane < 16) y[n0 + lane] = c[0] + bd2[n0 + lane];
}

// ---------------- predict transform ----------------
__global__ void predict_kernel(const float* __restrict__ y, float* __restrict__ pred) {
    int cell = blockIdx.x * 256 + threadIdx.x;
    if (cell >= GH_ * GW_) return;
    int gx = cell % GW_, gy = cell / GW_;
    const float* v = y + cell * 16;
    float* o = pred + cell * 16;
    float fx = (float)gx, fy = (float)gy;
    o[0]  = (sigmoidf(v[0]) + fx) * 16.0f;
    o[1]  = (sigmoidf(v[1]) + fy) * 16.0f;
    o[2]  = __expf(v[2]) * 60.0f * 16.0f;   // anc1.x = IMW/STRIDE
    o[3]  = __expf(v[3]) * 30.0f * 16.0f;   // anc1.y = IMW/2/STRIDE
    o[4]  = sigmoidf(v[4]);
    o[5]  = sigmoidf(v[5]);
    o[6]  = sigmoidf(v[6]);
    o[7]  = sigmoidf(v[7]);
    o[8]  = (v[8] + fx) * 16.0f;
    o[9]  = (v[9] + fy) * 16.0f;
    o[10] = __expf(v[10]) * 20.0f * 16.0f;  // anc2.x = IMH/2/STRIDE
    o[11] = __expf(v[11]) * 40.0f * 16.0f;  // anc2.y = IMH/STRIDE
    o[12] = v[12];
    o[13] = sigmoidf(v[13]);
    o[14] = sigmoidf(v[14]);
    o[15] = sigmoidf(v[15]);
}

// ---------------- boxes: conf threshold + class argmax ----------------
__global__ void boxes_kernel(const float* __restrict__ pred, float* __restrict__ ip) {
    int bx = blockIdx.x * 256 + threadIdx.x;
    if (bx >= NBOX) return;
    const float* p = pred + (bx >> 1) * 16 + (bx & 1) * 8;
    float conf = p[4];
    float m = (conf > 0.5f) ? 1.0f : 0.0f;
    float x = p[0] * m, y = p[1] * m, w = p[2] * m, h = p[3] * m, cf = conf * m;
    float c0 = p[5] * m, c1 = p[6] * m, c2 = p[7] * m;
    int ind = 0; float best = c0;
    if (c1 > best) { best = c1; ind = 1; }
    if (c2 > best) { best = c2; ind = 2; }
    float* o = ip + bx * 7;
    o[0] = x; o[1] = y; o[2] = w; o[3] = h; o[4] = cf;
    o[5] = (float)ind; o[6] = best;
}

// ---------------- single-block bitonic sort (descending conf, stable) --------
__global__ void sort_kernel(const float* __restrict__ ip, unsigned long long* keys,
                            int* __restrict__ order) {
    int tid = threadIdx.x;  // 1024 threads
    for (int i = tid; i < NPAD; i += 1024) {
        unsigned long long kk = 0ull;
        if (i < NBOX) {
            unsigned u = __float_as_uint(ip[i * 7 + 4]);
            u = (u & 0x80000000u) ? ~u : (u | 0x80000000u);  // total order
            kk = ((unsigned long long)u << 32) | (unsigned)(~(unsigned)i);
        }
        keys[i] = kk;
    }
    __syncthreads();
    for (unsigned k = 2; k <= NPAD; k <<= 1) {
        for (unsigned j = k >> 1; j > 0; j >>= 1) {
            for (unsigned i = tid; i < NPAD; i += 1024) {
                unsigned ixj = i ^ j;
                if (ixj > i) {
                    unsigned long long a = keys[i], b = keys[ixj];
                    bool up = ((i & k) == 0);
                    bool sw = up ? (a < b) : (a > b);   // descending overall
                    if (sw) { keys[i] = b; keys[ixj] = a; }
                }
            }
            __syncthreads();
        }
    }
    for (int i = tid; i < NBOX; i += 1024)
        order[i] = (int)(~(unsigned)(keys[i] & 0xFFFFFFFFull));
}

__global__ void gather_kernel(const float* __restrict__ ip, const int* __restrict__ order,
                              float* __restrict__ sp) {
    int i = blockIdx.x * 256 + threadIdx.x;
    if (i >= NBOX) return;
    int src = order[i];
    #pragma unroll
    for (int c = 0; c < 7; ++c) sp[i * 7 + c] = ip[src * 7 + c];
}

// ---------------- suppression bitmask matrix: 4800 rows x 150 words ----------
__global__ void supp_kernel(const float* __restrict__ sp, unsigned* __restrict__ supp) {
    int t = blockIdx.x * 256 + threadIdx.x;
    if (t >= NBOX * NWORD) return;
    int row = t / NWORD, w = t % NWORD;
    float x = sp[row * 7 + 0], y = sp[row * 7 + 1];
    float wd = sp[row * 7 + 2], hh = sp[row * 7 + 3], cls = sp[row * 7 + 5];
    float x1 = x - wd * 0.5f, x2 = x + wd * 0.5f;
    float y1 = y - hh * 0.5f, y2 = y + hh * 0.5f;
    float area = (x2 - x1 + 1.0f) * (y2 - y1 + 1.0f);
    unsigned bits = 0u;
    #pragma unroll 4
    for (int b = 0; b < 32; ++b) {
        int j = w * 32 + b;
        float xj = sp[j * 7 + 0], yj = sp[j * 7 + 1];
        float wj = sp[j * 7 + 2], hj = sp[j * 7 + 3], cj = sp[j * 7 + 5];
        float xj1 = xj - wj * 0.5f, xj2 = xj + wj * 0.5f;
        float yj1 = yj - hj * 0.5f, yj2 = yj + hj * 0.5f;
        float iw = fminf(x2, xj2) - fmaxf(x1, xj1) + 1.0f;
        iw = fminf(fmaxf(iw, 0.0f), 2000.0f);
        float ihh = fminf(y2, yj2) - fmaxf(y1, yj1) + 1.0f;
        ihh = fminf(fmaxf(ihh, 0.0f), 2000.0f);
        float inter = iw * ihh;
        float aj = (xj2 - xj1 + 1.0f) * (yj2 - yj1 + 1.0f);
        float iou = inter / (area + aj - inter);
        if (cls == cj && iou >= 0.4f) bits |= (1u << b);
    }
    supp[t] = bits;
}

// ---------------- sequential greedy NMS sweep (exact ref semantics) ----------
__global__ void nms_sweep(const float* __restrict__ sp, const unsigned* __restrict__ supp,
                          float* __restrict__ det) {
    __shared__ unsigned S[NWORD];
    __shared__ unsigned keepW[NWORD];
    __shared__ int flag;
    int tid = threadIdx.x;  // 256
    for (int w = tid; w < NWORD; w += 256) { S[w] = 0u; keepW[w] = 0u; }
    __syncthreads();
    for (int i = 0; i < NBOX; ++i) {
        if (tid == 0) {
            bool valid = (sp[i * 7 + 0] != 0.0f);
            bool suppressed = (S[i >> 5] >> (i & 31)) & 1u;
            flag = (valid && !suppressed) ? 1 : 0;
            if (flag) keepW[i >> 5] |= (1u << (i & 31));
        }
        __syncthreads();
        if (flag) {
            for (int w = tid; w < NWORD; w += 256) S[w] |= supp[i * NWORD + w];
        }
        __syncthreads();
    }
    for (int t = tid; t < NBOX * 7; t += 256) {
        int i = t / 7;
        float kf = ((keepW[i >> 5] >> (i & 31)) & 1u) ? 1.0f : 0.0f;
        det[t] = sp[t] * kf;
    }
}

// ---------------- host launch ----------------
extern "C" void kernel_launch(void* const* d_in, const int* in_sizes, int n_in,
                              void* d_out, int out_size, void* d_ws, size_t ws_size,
                              hipStream_t stream) {
    const float* img = (const float*)d_in[0];
    const float* w1  = (const float*)d_in[1];
    const float* b1  = (const float*)d_in[2];
    const float* w2  = (const float*)d_in[3];
    const float* b2  = (const float*)d_in[4];
    const float* w3  = (const float*)d_in[5];
    const float* b3  = (const float*)d_in[6];
    const float* wd1 = (const float*)d_in[7];
    const float* bd1 = (const float*)d_in[8];
    const float* wd2 = (const float*)d_in[9];
    const float* bd2 = (const float*)d_in[10];

    float* pred = (float*)d_out;            // 38400
    float* det  = (float*)d_out + NOUT;     // 4800*7

    char* ws = (char*)d_ws;
    size_t off = 0;
    auto carve = [&](size_t bytes) -> char* {
        char* p = ws + off;
        off = (off + bytes + 255) & ~(size_t)255;
        return p;
    };
    float* a1    = (float*)carve((size_t)C1H * C1W * 32 * 4);
    float* p1    = (float*)carve((size_t)P1H * P1W * 32 * 4);
    float* a2    = (float*)carve((size_t)C2H * C2W * 64 * 4);
    float* p2    = (float*)carve((size_t)P2H * P2W * 64 * 4);
    _Float16* xh = (_Float16*)carve((size_t)FLATK * 2);
    float* h1    = (float*)carve((size_t)HID * 4);
    _Float16* h1h = (_Float16*)carve((size_t)HID * 2);
    float* y     = (float*)carve((size_t)NOUT * 4);
    float* ip    = (float*)carve((size_t)NBOX * 7 * 4);
    unsigned long long* keys = (unsigned long long*)carve((size_t)NPAD * 8);
    int*   order = (int*)carve((size_t)NBOX * 4);
    float* sp    = (float*)carve((size_t)NBOX * 7 * 4);
    unsigned* supp = (unsigned*)carve((size_t)NBOX * NWORD * 4);

    init_h1<<<1, 256, 0, stream>>>(h1);
    conv1_kernel<<<(C1H * C1W * 32 + 255) / 256, 256, 0, stream>>>(img, w1, b1, a1);
    maxpool_kernel<<<(P1H * P1W * 32 + 255) / 256, 256, 0, stream>>>(a1, p1, P1H, P1W, C1W, 32, 5);
    conv2_kernel<<<(C2H * C2W * 64 + 255) / 256, 256, 0, stream>>>(p1, w2, b2, a2);
    maxpool_kernel<<<(P2H * P2W * 64 + 255) / 256, 256, 0, stream>>>(a2, p2, P2H, P2W, C2W, 64, 6);
    conv3_kernel<<<(FLATK + 255) / 256, 256, 0, stream>>>(p2, w3, b3, xh);
    dense1_wmma<<<dim3(16, 16), 256, 0, stream>>>(xh, wd1, h1);
    bias_relu_h1<<<1, 256, 0, stream>>>(h1, bd1, h1h);
    dense2_wmma<<<NOUT / 16 / 8, 256, 0, stream>>>(h1h, wd2, bd2, y);
    predict_kernel<<<(GH_ * GW_ + 255) / 256, 256, 0, stream>>>(y, pred);
    boxes_kernel<<<(NBOX + 255) / 256, 256, 0, stream>>>(pred, ip);
    sort_kernel<<<1, 1024, 0, stream>>>(ip, keys, order);
    gather_kernel<<<(NBOX + 255) / 256, 256, 0, stream>>>(ip, order, sp);
    supp_kernel<<<(NBOX * NWORD + 255) / 256, 256, 0, stream>>>(sp, supp);
    nms_sweep<<<1, 256, 0, stream>>>(sp, supp, det);
}
